// OneLayerRNN_82403242541078
// MI455X (gfx1250) — matrix operand order repaired and checked by reading, not compiled, not run on hardware
//
#include <hip/hip_runtime.h>
#include <hip/hip_bf16.h>

// ---------------------------------------------------------------------------
// Particle-filter RNN on MI455X (gfx1250, wave32, WMMA f16->f32).
//   k0: convert W_ih / W_hh / fc_w f32 -> f16 into workspace (1.31 MB).
//   k1: parallel WMMA GEMM  preact[b,t,p,:] = x[b,p,t,:] @ W_ih^T + b_ih + b_hh,
//       stored directly into d_out (later overwritten in-place by tanh output).
//   k2: 16 persistent workgroups (one per batch b), 16 waves each, loop t=0..1023:
//       resample (Gumbel top-k), gather, h_res @ W_hh^T (+preact C-accum),
//       +noise, tanh, store to d_out, fc GEMM, ||mu||^2 via ds_add_f32, softmax.
// ---------------------------------------------------------------------------

typedef __attribute__((ext_vector_type(16))) _Float16 v16h;
typedef __attribute__((ext_vector_type(8)))  _Float16 v8h;
typedef __attribute__((ext_vector_type(8)))  float    v8f;
typedef __attribute__((ext_vector_type(4)))  float    v4f;

#define BB 16
#define PP 10
#define TT 1024
#define DI 512
#define HH 512
#define DY 256
#define LDA 520   // LDS row stride in halves: 520*2B = 1040B -> rows 4 banks apart

// ---------------- deterministic counter-based RNG --------------------------
__device__ __forceinline__ unsigned hash_u32(unsigned long long x) {
  x ^= x >> 33; x *= 0xff51afd7ed558ccdULL;
  x ^= x >> 33; x *= 0xc4ceb9fe1a85ec53ULL;
  x ^= x >> 33;
  return (unsigned)x;
}
__device__ __forceinline__ float u01(unsigned h) {
  return (float)(h >> 8) * (1.0f / 16777216.0f) + (0.5f / 16777216.0f); // (0,1)
}

// ---------------- WMMA fragment loaders (contiguous k order, same in A and B) ---------------
// A 16x32 f16: lane l: m=l&15, kh=l>>4; elem e -> K=kh*16+e (0..15).
// Two 16B LDS reads.
__device__ __forceinline__ v16h load_a_frag(const _Float16* A, int lane, int ktile) {
  const int m = lane & 15, kh = lane >> 4;
  const _Float16* p = A + m * LDA + ktile * 32 + kh * 16;
  v8h lo = *(const v8h*)(p);
  v8h hi = *(const v8h*)(p + 8);
  v16h r;
#pragma unroll
  for (int i = 0; i < 8; ++i) { r[i] = lo[i]; r[i + 8] = hi[i]; }
  return r;
}
// B 32x16 f16 for D = A @ W^T: B[k,n] = W[n,k], W row-major [N][K] f16.
// lane l: n=l&15, kb=l>>4; elem e -> K = ktile*32 + kb*16 + e (32B contiguous).
__device__ __forceinline__ v16h load_b_frag(const _Float16* W, int lane, int n0,
                                            int ktile, int ldk) {
  const int n = n0 + (lane & 15), kb = lane >> 4;
  return *(const v16h*)(W + (size_t)n * ldk + ktile * 32 + kb * 16);
}

// ---------------- k0: weight conversion f32 -> f16 -------------------------
__global__ void k0_convert(const float* __restrict__ Wih, const float* __restrict__ Whh,
                           const float* __restrict__ fcw, _Float16* __restrict__ dst) {
  int i = blockIdx.x * 256 + threadIdx.x;
  if (i < HH * DI)                   dst[i] = (_Float16)Wih[i];
  else if (i < 2 * HH * DI)          dst[i] = (_Float16)Whh[i - HH * DI];
  else if (i < 2 * HH * DI + DY * HH) dst[i] = (_Float16)fcw[i - 2 * HH * DI];
}

// ---------------- k1: parallel input GEMM into d_out -----------------------
// grid = (T/16, B*P), 256 threads (8 waves). Each WG: 16 t-rows x full N=512.
__global__ __launch_bounds__(256) void k1_input_gemm(
    const float* __restrict__ input, const float* __restrict__ b_ih,
    const float* __restrict__ b_hh, const _Float16* __restrict__ Wih16,
    float* __restrict__ out) {
  const int ttile = blockIdx.x, bp = blockIdx.y;
  const int b = bp / PP, p = bp % PP;
  __shared__ _Float16 As[16 * LDA];

  // cooperative load 16x512 f32 (contiguous in t for fixed (b,p)) -> f16 LDS
  const float* src = input + (((size_t)bp) * TT + (size_t)ttile * 16) * DI;
  for (int i = threadIdx.x; i < 16 * DI / 4; i += 256) {
    const int row = i >> 7, col = (i << 2) & (DI - 1);
    v4f v = *(const v4f*)(src + row * DI + col);
    _Float16* d = As + row * LDA + col;
    d[0] = (_Float16)v[0]; d[1] = (_Float16)v[1];
    d[2] = (_Float16)v[2]; d[3] = (_Float16)v[3];
  }
  __syncthreads();

  const int wave = threadIdx.x >> 5, lane = threadIdx.x & 31;
#pragma unroll
  for (int nt = 0; nt < 4; ++nt) {
    const int n0 = wave * 64 + nt * 16;
    v8f c = {};
#pragma unroll 4
    for (int kt = 0; kt < DI / 32; ++kt) {
      v16h a  = load_a_frag(As, lane, kt);
      v16h bf = load_b_frag(Wih16, lane, n0, kt, DI);
      if (kt + 1 < DI / 32)
        __builtin_prefetch((const void*)(Wih16 + (size_t)n0 * DI + (kt + 1) * 32), 0, 1);
      c = __builtin_amdgcn_wmma_f32_16x16x32_f16(false, a, false, bf,
                                                 (short)0, c, false, false);
    }
    const int n = n0 + (lane & 15);
    const float bias = b_ih[n] + b_hh[n];
    const int mb = (lane >> 4) * 8;
#pragma unroll
    for (int r = 0; r < 8; ++r) {
      const int t = ttile * 16 + mb + r;
      out[(((size_t)b * TT + t) * PP + p) * HH + n] = c[r] + bias;
    }
  }
}

// ---------------- k2: sequential particle-filter recurrence ----------------
// grid = 16 (one WG per batch), 512 threads = 16 waves.
__global__ __launch_bounds__(512) void k2_seq(
    const float* __restrict__ target, const _Float16* __restrict__ Whh16,
    const _Float16* __restrict__ fcw16, const float* __restrict__ fcb,
    const float* __restrict__ rw0, float* __restrict__ out) {
  const int b = blockIdx.x;
  __shared__ _Float16 hA[16 * LDA];   // current hidden state (f16), 16 rows
  __shared__ _Float16 hR[16 * LDA];   // resampled hidden state
  __shared__ float    wsm[16];        // normalized particle weights
  __shared__ int      It[16];         // resampling indices
  __shared__ float    sumsq[16];      // per-particle ||mu||^2

  const int tid = threadIdx.x, lane = tid & 31, wave = tid >> 5;

  for (int i = tid; i < 16 * LDA; i += 512) { hA[i] = (_Float16)0.f; hR[i] = (_Float16)0.f; }
  if (tid == 0) {  // w0 = softmax(rw0[b,:])
    float mx = -1e30f;
    for (int p = 0; p < PP; ++p) mx = fmaxf(mx, rw0[b * PP + p]);
    float e[PP], s = 0.f;
    for (int p = 0; p < PP; ++p) { e[p] = __expf(rw0[b * PP + p] - mx); s += e[p]; }
    for (int p = 0; p < PP; ++p) wsm[p] = e[p] / s;
  }
  __syncthreads();

  for (int t = 0; t < TT; ++t) {
    // --- 1. Gumbel top-k resampling indices (P=10, scalar work on thread 0)
    if (tid == 0) {
      float key[PP]; int idx[PP];
      for (int p = 0; p < PP; ++p) {
        unsigned h = hash_u32((((unsigned long long)(b * PP + p)) << 32) ^
                              (unsigned long long)(2 * t + 1) ^ 0x9E3779B97F4A7C15ULL);
        float g = -__logf(-__logf(u01(h)));
        key[p] = __logf(fmaxf(wsm[p], 1e-30f)) + g;
        idx[p] = p;
      }
      for (int i = 0; i < PP; ++i) {       // selection sort, descending keys
        int best = i;
        for (int j = i + 1; j < PP; ++j) if (key[idx[j]] > key[idx[best]]) best = j;
        int tmp = idx[i]; idx[i] = idx[best]; idx[best] = tmp;
        It[i] = idx[i];
      }
    }
    if (tid < 16) sumsq[tid] = 0.f;
    __syncthreads();

    // --- 2. gather resampled particles: hR[p] = hA[It[p]]  (rows 10..15 stay 0)
    for (int i = tid; i < PP * (HH / 8); i += 512) {
      const int p = i / (HH / 8), col = (i % (HH / 8)) * 8;
      *(v8h*)(hR + p * LDA + col) = *(const v8h*)(hA + It[p] * LDA + col);
    }
    __syncthreads();

    // --- 3. act = preact(d_out) + hR @ W_hh^T + noise;  hy = tanh(act)
    float hyreg[2][8];
#pragma unroll
    for (int nt = 0; nt < 2; ++nt) {
      const int n0 = wave * 32 + nt * 16;
      const int n  = n0 + (lane & 15);
      const int mb = (lane >> 4) * 8;
      v8f c;
#pragma unroll
      for (int r = 0; r < 8; ++r) {
        const int p = mb + r;
        c[r] = (p < PP) ? out[(((size_t)b * TT + t) * PP + p) * HH + n] : 0.f;
      }
#pragma unroll 4
      for (int kt = 0; kt < HH / 32; ++kt) {
        v16h a  = load_a_frag(hR, lane, kt);
        v16h bf = load_b_frag(Whh16, lane, n0, kt, HH);
        c = __builtin_amdgcn_wmma_f32_16x16x32_f16(false, a, false, bf,
                                                   (short)0, c, false, false);
      }
#pragma unroll
      for (int r = 0; r < 8; ++r) {
        const int m = mb + r;
        const unsigned long long ctr =
            (((unsigned long long)(b * TT + t)) << 32) | (unsigned)(m * HH + n);
        const float un1 = u01(hash_u32(ctr ^ 0xD6E8FEB86659FD93ULL));
        const float un2 = u01(hash_u32(ctr ^ 0xA5A5A5A5DEADBEEFULL));
        const float z = __fsqrt_rn(-2.f * __logf(un1)) * __cosf(6.28318530718f * un2);
        hyreg[nt][r] = tanhf(c[r] + 0.70710678f * z);   // sqrt(SIGMA_H)=sqrt(0.5)
      }
    }
    // store hy: f16 -> hA (all 16 rows, pad rows never gathered), f32 -> d_out
#pragma unroll
    for (int nt = 0; nt < 2; ++nt) {
      const int n  = wave * 32 + nt * 16 + (lane & 15);
      const int mb = (lane >> 4) * 8;
#pragma unroll
      for (int r = 0; r < 8; ++r) {
        const int m = mb + r;
        const float hy = hyreg[nt][r];
        hA[m * LDA + n] = (_Float16)hy;
        if (m < PP) {
          out[(((size_t)b * TT + t) * PP + m) * HH + n] = hy;
          if (t == TT - 1)
            out[(size_t)BB * TT * PP * HH + ((size_t)b * PP + m) * HH + n] = hy;
        }
      }
    }
    __syncthreads();

    // --- 4. pred = hy @ fc_w^T + fc_b ;  sumsq[p] += (y - pred)^2
    {
      const int n0 = wave * 16;                 // 16 waves x 16 cols = Dy=256
      const int n  = n0 + (lane & 15);
      const int mb = (lane >> 4) * 8;
      const float fb = fcb[n];
      v8f c;
#pragma unroll
      for (int r = 0; r < 8; ++r) c[r] = fb;
#pragma unroll 4
      for (int kt = 0; kt < HH / 32; ++kt) {
        v16h a  = load_a_frag(hA, lane, kt);
        v16h bf = load_b_frag(fcw16, lane, n0, kt, HH);
        c = __builtin_amdgcn_wmma_f32_16x16x32_f16(false, a, false, bf,
                                                   (short)0, c, false, false);
      }
#pragma unroll
      for (int r = 0; r < 8; ++r) {
        const int p = mb + r;
        if (p < PP) {
          const float y  = target[(((size_t)b * PP + p) * TT + t) * DY + n];
          const float mu = y - c[r];
          atomicAdd(&sumsq[p], mu * mu);        // ds_add_f32
        }
      }
    }
    __syncthreads();

    // --- 5. w = softmax(-sumsq / (2*SIGMA_Y)) = softmax(-sumsq)
    if (tid == 0) {
      float lw[PP], mx = -1e30f;
      for (int p = 0; p < PP; ++p) { lw[p] = -sumsq[p]; mx = fmaxf(mx, lw[p]); }
      float s = 0.f;
      for (int p = 0; p < PP; ++p) { lw[p] = __expf(lw[p] - mx); s += lw[p]; }
      for (int p = 0; p < PP; ++p) wsm[p] = lw[p] / s;
    }
    __syncthreads();
  }
}

// ---------------------------------------------------------------------------
extern "C" void kernel_launch(void* const* d_in, const int* in_sizes, int n_in,
                              void* d_out, int out_size, void* d_ws, size_t ws_size,
                              hipStream_t stream) {
  const float* input  = (const float*)d_in[0];
  const float* target = (const float*)d_in[1];
  const float* W_ih   = (const float*)d_in[2];
  const float* b_ih   = (const float*)d_in[3];
  const float* W_hh   = (const float*)d_in[4];
  const float* b_hh   = (const float*)d_in[5];
  const float* fc_w   = (const float*)d_in[6];
  const float* fc_b   = (const float*)d_in[7];
  const float* rw0    = (const float*)d_in[8];
  float* out = (float*)d_out;

  _Float16* w16   = (_Float16*)d_ws;            // 655,360 halves = 1.31 MB
  _Float16* Wih16 = w16;
  _Float16* Whh16 = w16 + HH * DI;
  _Float16* fcw16 = w16 + 2 * HH * DI;

  const int conv_n = 2 * HH * DI + DY * HH;
  k0_convert<<<(conv_n + 255) / 256, 256, 0, stream>>>(W_ih, W_hh, fc_w, w16);

  dim3 g1(TT / 16, BB * PP);                    // 64 x 160 workgroups
  k1_input_gemm<<<g1, 256, 0, stream>>>(input, b_ih, b_hh, Wih16, out);

  k2_seq<<<BB, 512, 0, stream>>>(target, Whh16, fcw16, fc_b, rw0, out);
}
